// MultiHeadAttention_32177894982271
// MI455X (gfx1250) — matrix operand order repaired and checked by reading, not compiled, not run on hardware
//
#include <hip/hip_runtime.h>
#include <stdint.h>

// Transformer-XL attention for MI455X (gfx1250), bf16 WMMA pipeline.
// B=4, S=1024, MEM=1024, T=2048, D=1024, H=16, DK=64.
//
//   0) one-shot f32->bf16 conversion of all operands
//   1) LDS-staged NT GEMMs (128x128 block tile, double-buffered via
//      GLOBAL_LOAD_ASYNC_TO_LDS_B128, XOR-swizzled LDS, pure
//      v_wmma_f32_16x16x32_bf16 inner loop) with fused epilogues
//   2) fused flash attention: async-copy double-buffered K/V LDS staging,
//      banded WMMA rel-pos term with wave-private LDS diagonal gather,
//      register-resident P -> B-fragment handoff
//   3) output projection GEMM -> f32

#define B_   4
#define S_   1024
#define MEM_ 1024
#define T_   2048
#define D_   1024
#define H_   16
#define DK_  64

typedef __attribute__((ext_vector_type(16))) __bf16 v16bf;
typedef __attribute__((ext_vector_type(8)))  float  v8f;

union FragU {
  unsigned u[8];
  v16bf    v;
};

__device__ __forceinline__ v8f vzero() {
  v8f z;
#pragma unroll
  for (int i = 0; i < 8; ++i) z[i] = 0.0f;
  return z;
}

__device__ __forceinline__ unsigned short f2bf(float f) {
  unsigned x = __float_as_uint(f);
  x += 0x7FFFu + ((x >> 16) & 1u);   // round-to-nearest-even
  return (unsigned short)(x >> 16);
}
__device__ __forceinline__ unsigned pk2bf(float lo, float hi) {
  return (unsigned)f2bf(lo) | ((unsigned)f2bf(hi) << 16);
}

// ---------------------------------------------------------------------------
// CDNA5 async global->LDS copy (bypasses VGPRs, tracked by ASYNCcnt).
// Generic pointers to __shared__ carry the wave-relative LDS offset in their
// low 32 bits (LDS aperture base occupies bits [63:32]), which is exactly the
// VDST operand the instruction expects.
// ---------------------------------------------------------------------------
__device__ __forceinline__ void async_ld16(const void* g, void* l) {
  asm volatile("global_load_async_to_lds_b128 %0, %1, off"
               :: "v"((unsigned)(uintptr_t)l),
                  "v"((unsigned long long)(uintptr_t)g)
               : "memory");
}
__device__ __forceinline__ void wait_async0() {
  asm volatile("s_wait_asynccnt 0" ::: "memory");
}

// ---------------------------------------------------------------------------
// One-shot f32 -> bf16 conversion, 8 elements/thread.
// ---------------------------------------------------------------------------
__global__ __launch_bounds__(256) void cvt_bf16_kernel(
    const float* __restrict__ in, unsigned short* __restrict__ out, int n8) {
  const int idx = blockIdx.x * 256 + threadIdx.x;
  if (idx >= n8) return;
  const float4* p = (const float4*)in + (size_t)idx * 2;
  float4 a = p[0];
  float4 b = p[1];
  uint4 w;
  w.x = pk2bf(a.x, a.y);
  w.y = pk2bf(a.z, a.w);
  w.z = pk2bf(b.x, b.y);
  w.w = pk2bf(b.z, b.w);
  ((uint4*)out)[idx] = w;
}

// ---------------------------------------------------------------------------
// XOR-swizzled LDS addressing (16B granules; conflict-free b128 r/w).
//   sw4: rows of 32 bf16 (4 granules)   sw8: rows of 64 bf16 (8 granules)
// ---------------------------------------------------------------------------
__device__ __forceinline__ int sw4(int row, int g) {
  return (row * 4 + (g ^ ((row >> 2) & 3))) * 8;   // element offset
}
__device__ __forceinline__ int sw8(int row, int g) {
  return (row * 8 + (g ^ ((row >> 1) & 7))) * 8;   // element offset
}

// ---------------------------------------------------------------------------
// WMMA fragment layouts (wave32, V_WMMA_F32_16X16X32_BF16).
// A (16Mx32K): lane&15=row; per-lane K = {8h..8h+7} U {16+8h..16+8h+7}.
// B (32Kx16N): lane&15=col; per-lane K = {8h..8h+7} U {16+8h..16+8h+7}.
// C/D (16x16 f32): lane&15=col, VGPR r -> row r + 8h.
// ---------------------------------------------------------------------------
__device__ __forceinline__ FragU mkfrag(uint4 a, uint4 b) {
  FragU f;
  f.u[0] = a.x; f.u[1] = a.y; f.u[2] = a.z; f.u[3] = a.w;
  f.u[4] = b.x; f.u[5] = b.y; f.u[6] = b.z; f.u[7] = b.w;
  return f;
}
__device__ __forceinline__ FragU load_fragA_g(const unsigned short* base, int row,
                                              int ld, int k0, int h) {
  const unsigned short* p = base + (size_t)row * ld + k0 + 8 * h;
  return mkfrag(*(const uint4*)(p), *(const uint4*)(p + 16));
}
__device__ __forceinline__ FragU load_fragB_g(const unsigned short* base, int row,
                                              int ld, int k0, int h) {
  const unsigned short* p = base + (size_t)row * ld + k0 + 8 * h;
  return mkfrag(*(const uint4*)(p), *(const uint4*)(p + 16));
}
__device__ __forceinline__ FragU fragA_lds4(const unsigned short* st, int row, int h) {
  return mkfrag(*(const uint4*)(st + sw4(row, h)),
                *(const uint4*)(st + sw4(row, 2 + h)));
}
__device__ __forceinline__ FragU fragB_lds4(const unsigned short* st, int row, int h) {
  return mkfrag(*(const uint4*)(st + sw4(row, h)),
                *(const uint4*)(st + sw4(row, 2 + h)));
}
__device__ __forceinline__ FragU fragA_lds8(const unsigned short* st, int row,
                                            int half, int h) {
  return mkfrag(*(const uint4*)(st + sw8(row, half * 4 + h)),
                *(const uint4*)(st + sw8(row, half * 4 + 2 + h)));
}

__device__ __forceinline__ v8f wmma_bf16(const FragU& a, const FragU& b, v8f c) {
  return __builtin_amdgcn_wmma_f32_16x16x32_bf16(false, a.v, false, b.v,
                                                 (short)0, c, false, false);
}

// ---------------------------------------------------------------------------
// LDS-staged NT GEMM: Y[M,N] = X[M,K] * W[N,K]^T, K = N = D_ = 1024, bf16 in.
// Block = 256 threads (8 waves), block tile 128x128, wave tile 32x64.
// Double-buffered A/B k-slices staged with async global->LDS copies.
// ---------------------------------------------------------------------------
enum { EP_QUQV = 0, EP_K = 1, EP_V = 2, EP_R = 3, EP_OUT = 4 };

template <int MODE>
__global__ __launch_bounds__(256) void gemm_nt_kernel(
    const unsigned short* __restrict__ A, const unsigned short* __restrict__ W,
    const float* __restrict__ ubias, const float* __restrict__ vbias,
    unsigned short* __restrict__ out0, unsigned short* __restrict__ out1,
    float* __restrict__ outf) {
  __shared__ unsigned short Ast[2][128 * 32];   // 2 x 8KB
  __shared__ unsigned short Bst[2][128 * 32];   // 2 x 8KB

  const int tid  = threadIdx.x;
  const int lane = tid & 31;
  const int wave = tid >> 5;
  const int di   = lane & 15;
  const int h    = lane >> 4;
  const int wm   = wave & 3;        // 4 M sub-tiles of 32
  const int wn   = wave >> 2;       // 2 N sub-tiles of 64
  const int m_blk = blockIdx.y * 128;
  const int n_blk = blockIdx.x * 128;

  // staging: thread -> (row, granule pair) of the 128x32 slice
  const int srow = tid >> 1;             // 0..127
  const int sg0  = (tid & 1) * 2;        // granule 0 or 2
  const unsigned short* Ag = A + (size_t)(m_blk + srow) * D_ + sg0 * 8;
  const unsigned short* Wg = W + (size_t)(n_blk + srow) * D_ + sg0 * 8;

  // stage k0 = 0 (async copies, wait, barrier)
  async_ld16(Ag,      &Ast[0][sw4(srow, sg0)]);
  async_ld16(Ag + 8,  &Ast[0][sw4(srow, sg0 + 1)]);
  async_ld16(Wg,      &Bst[0][sw4(srow, sg0)]);
  async_ld16(Wg + 8,  &Bst[0][sw4(srow, sg0 + 1)]);
  wait_async0();
  __syncthreads();

  v8f c[2][4];
#pragma unroll
  for (int mi = 0; mi < 2; ++mi)
#pragma unroll
    for (int ni = 0; ni < 4; ++ni) c[mi][ni] = vzero();

  int p = 0;
  for (int k0 = 0; k0 < D_; k0 += 32) {
    const bool more = (k0 + 32) < D_;
    if (more) {                       // async-stage next slice during compute
      async_ld16(Ag + k0 + 32, &Ast[p ^ 1][sw4(srow, sg0)]);
      async_ld16(Ag + k0 + 40, &Ast[p ^ 1][sw4(srow, sg0 + 1)]);
      async_ld16(Wg + k0 + 32, &Bst[p ^ 1][sw4(srow, sg0)]);
      async_ld16(Wg + k0 + 40, &Bst[p ^ 1][sw4(srow, sg0 + 1)]);
    }
    FragU a[2], b[4];
#pragma unroll
    for (int mi = 0; mi < 2; ++mi)
      a[mi] = fragA_lds4(Ast[p], wm * 32 + mi * 16 + di, h);
#pragma unroll
    for (int ni = 0; ni < 4; ++ni)
      b[ni] = fragB_lds4(Bst[p], wn * 64 + ni * 16 + di, h);
#pragma unroll
    for (int mi = 0; mi < 2; ++mi)
#pragma unroll
      for (int ni = 0; ni < 4; ++ni)
        c[mi][ni] = wmma_bf16(a[mi], b[ni], c[mi][ni]);
    if (more) {
      wait_async0();                  // our chunks landed in LDS
      __syncthreads();                // everyone's chunks landed
      p ^= 1;
    }
  }

#pragma unroll
  for (int mi = 0; mi < 2; ++mi)
#pragma unroll
    for (int ni = 0; ni < 4; ++ni)
#pragma unroll
      for (int r = 0; r < 8; ++r) {
        const int mg = m_blk + wm * 32 + mi * 16 + r + 8 * h;
        const int ng = n_blk + wn * 64 + ni * 16 + di;
        const float val = c[mi][ni][r];
        if (MODE == EP_QUQV) {
          const int hh = ng >> 6, dk = ng & 63;
          const int bb = mg >> 10, s = mg & (S_ - 1);
          const size_t idx = ((((size_t)bb * H_ + hh) * S_) + s) * DK_ + dk;
          // fold 1/sqrt(DK) into q so attention scores need no rescale
          out0[idx] = f2bf((val + ubias[hh * DK_ + dk]) * 0.125f);
          out1[idx] = f2bf((val + vbias[hh * DK_ + dk]) * 0.125f);
        } else if (MODE == EP_K) {
          const int hh = ng >> 6, dk = ng & 63;
          const int bb = mg >> 11, t = mg & (T_ - 1);
          out0[((((size_t)bb * H_ + hh) * T_) + t) * DK_ + dk] = f2bf(val);
        } else if (MODE == EP_V) {
          const int hh = ng >> 6, dk = ng & 63;
          const int bb = mg >> 11, t = mg & (T_ - 1);
          out0[((((size_t)bb * H_ + hh) * DK_) + dk) * T_ + t] = f2bf(val);  // V^T
        } else if (MODE == EP_R) {
          const int hh = ng >> 6, dk = ng & 63;
          out0[(((size_t)hh * T_) + mg) * DK_ + dk] = f2bf(val);
        } else {  // EP_OUT
          outf[(size_t)mg * D_ + ng] = val;
        }
      }
}

// ---------------------------------------------------------------------------
// Fused flash attention with TXL relative position.
// Block = 256 threads (8 waves); each wave owns 16 queries, full DK=64.
// K (32x64) and V^T (64x32) tiles async-staged to swizzled LDS, double
// buffered with one barrier per iteration.
// ---------------------------------------------------------------------------
__global__ __launch_bounds__(256) void attn_kernel(
    const unsigned short* __restrict__ QU, const unsigned short* __restrict__ QV,
    const unsigned short* __restrict__ KH, const unsigned short* __restrict__ VHT,
    const unsigned short* __restrict__ RH, unsigned short* __restrict__ O) {
  __shared__ unsigned short Kst[2][32 * 64];   // 2 x 4KB, sw8 rows
  __shared__ unsigned short Vst[2][64 * 32];   // 2 x 4KB, sw4 rows
  __shared__ float ldsbd[8][48 * 17];          // per-wave bd band

  const int tid  = threadIdx.x;
  const int lane = tid & 31;
  const int wave = tid >> 5;
  const int di   = lane & 15;
  const int h    = lane >> 4;
  const int bh   = blockIdx.y;
  const int b    = bh >> 4;
  const int hh   = bh & 15;
  const int i0   = blockIdx.x * 128 + wave * 16;

  const unsigned short* QUb = QU + (size_t)bh * S_ * DK_;
  const unsigned short* QVb = QV + (size_t)bh * S_ * DK_;
  const unsigned short* KHb = KH + (size_t)bh * T_ * DK_;
  const unsigned short* Vb  = VHT + (size_t)bh * DK_ * T_;
  const unsigned short* Rb  = RH + (size_t)hh * T_ * DK_;

  // staging: thread -> granule of the K tile (32 rows x 8) / V tile (64 x 4)
  const int krow = tid >> 3, kg = tid & 7;
  const int vrow = tid >> 2, vg = tid & 3;

  FragU qu[2], qv[2];
  qu[0] = load_fragB_g(QUb, i0 + di, DK_, 0, h);
  qu[1] = load_fragB_g(QUb, i0 + di, DK_, 32, h);
  qv[0] = load_fragB_g(QVb, i0 + di, DK_, 0, h);
  qv[1] = load_fragB_g(QVb, i0 + di, DK_, 32, h);

  // block-uniform key range (early waves do a few inert fully-masked tiles)
  const int j_end = min(T_, blockIdx.x * 128 + 112 + 16 + MEM_);

  // stage tile j0 = 0
  async_ld16(KHb + (size_t)krow * DK_ + kg * 8, &Kst[0][sw8(krow, kg)]);
  async_ld16(Vb + (size_t)vrow * T_ + vg * 8,   &Vst[0][sw4(vrow, vg)]);
  wait_async0();
  __syncthreads();

  v8f oc[4];
#pragma unroll
  for (int dt = 0; dt < 4; ++dt) oc[dt] = vzero();
  float m_run = -3.0e38f, l_run = 0.0f;

  int p = 0;
  for (int j0 = 0; j0 < j_end; j0 += 32) {
    const bool more = (j0 + 32) < j_end;
    if (more) {   // async-stage next K/V tile while we compute this one
      async_ld16(KHb + (size_t)(j0 + 32 + krow) * DK_ + kg * 8,
                 &Kst[p ^ 1][sw8(krow, kg)]);
      async_ld16(Vb + (size_t)vrow * T_ + (j0 + 32) + vg * 8,
                 &Vst[p ^ 1][sw4(vrow, vg)]);
    }

    // R band fragment loads (per-wave, direct from global/L2)
    const int p0 = i0 + j0 + (S_ - 1);
    FragU ra[3][2];
#pragma unroll
    for (int bt = 0; bt < 3; ++bt) {
      const int row = (p0 + bt * 16 + di) & (T_ - 1);
      ra[bt][0] = load_fragA_g(Rb, row, DK_, 0, h);
      ra[bt][1] = load_fragA_g(Rb, row, DK_, 32, h);
    }
    __builtin_prefetch(Rb + (size_t)((p0 + 48 + di) & (T_ - 1)) * DK_, 0, 3);
    __builtin_prefetch(Rb + (size_t)((p0 + 64 + di) & (T_ - 1)) * DK_, 0, 3);

    // ---- content term: S^T tiles (keys x queries) from staged K ----
    v8f sc[2];
#pragma unroll
    for (int jt = 0; jt < 2; ++jt) {
      FragU ka0 = fragA_lds8(Kst[p], jt * 16 + di, 0, h);
      FragU ka1 = fragA_lds8(Kst[p], jt * 16 + di, 1, h);
      v8f z = vzero();
      z = wmma_bf16(ka0, qu[0], z);
      sc[jt] = wmma_bf16(ka1, qu[1], z);
    }

    // ---- position term: 3 banded tiles -> wave-private LDS ----
#pragma unroll
    for (int bt = 0; bt < 3; ++bt) {
      v8f d = vzero();
      d = wmma_bf16(ra[bt][0], qv[0], d);
      d = wmma_bf16(ra[bt][1], qv[1], d);
#pragma unroll
      for (int r = 0; r < 8; ++r)
        ldsbd[wave][(bt * 16 + r + 8 * h) * 17 + di] = d[r];
    }
    asm volatile("s_wait_dscnt 0" ::: "memory");

    // ---- assemble scores (scale pre-folded into q), online softmax ----
    float pv[16];
    float tmax = -3.0e38f;
    if (j0 + 31 > i0 + MEM_) {          // masked boundary tiles (uniform)
#pragma unroll
      for (int jt = 0; jt < 2; ++jt)
#pragma unroll
        for (int r = 0; r < 8; ++r) {
          const int djl = jt * 16 + r + 8 * h;
          float s = sc[jt][r] + ldsbd[wave][(di + djl) * 17 + di];
          if (j0 + djl > i0 + di + MEM_) s = -3.0e38f;
          pv[jt * 8 + r] = s;
          tmax = fmaxf(tmax, s);
        }
    } else {                            // fast path: no mask compares
#pragma unroll
      for (int jt = 0; jt < 2; ++jt)
#pragma unroll
        for (int r = 0; r < 8; ++r) {
          const int djl = jt * 16 + r + 8 * h;
          const float s = sc[jt][r] + ldsbd[wave][(di + djl) * 17 + di];
          pv[jt * 8 + r] = s;
          tmax = fmaxf(tmax, s);
        }
    }
    tmax = fmaxf(tmax, __shfl_xor(tmax, 16, 32));
    const float m_new = fmaxf(m_run, tmax);
    const float scale = __expf(m_run - m_new);
    float lsum = 0.0f;
#pragma unroll
    for (int q = 0; q < 16; ++q) {
      pv[q] = __expf(pv[q] - m_new);
      lsum += pv[q];
    }
    lsum += __shfl_xor(lsum, 16, 32);
    l_run = l_run * scale + lsum;
    m_run = m_new;
#pragma unroll
    for (int dt = 0; dt < 4; ++dt)
#pragma unroll
      for (int r = 0; r < 8; ++r) oc[dt][r] *= scale;

    // ---- P -> B-fragment (32 keys x 16 queries) built in-lane from own P values ----
    unsigned px[4], py[4], pxo[4], pyo[4];
#pragma unroll
    for (int q = 0; q < 4; ++q) {
      px[q] = pk2bf(pv[2 * q], pv[2 * q + 1]);
      py[q] = pk2bf(pv[8 + 2 * q], pv[8 + 2 * q + 1]);
    }
#pragma unroll
    for (int q = 0; q < 4; ++q) {
      pxo[q] = __shfl_xor(px[q], 16, 32);
      pyo[q] = __shfl_xor(py[q], 16, 32);
    }
    FragU pb;
#pragma unroll
    for (int q = 0; q < 4; ++q) {
      pb.u[q]     = px[q];
      pb.u[4 + q] = py[q];
    }

    // ---- O^T += V^T_tile * P from staged V ----
#pragma unroll
    for (int dt = 0; dt < 4; ++dt) {
      FragU va = fragA_lds4(Vst[p], dt * 16 + di, h);
      oc[dt] = wmma_bf16(va, pb, oc[dt]);
    }

    // ---- commit next stage: async writes done, then block-wide barrier ----
    if (more) {
      wait_async0();
      __syncthreads();
      p ^= 1;
    }
  }

  // ---- normalize & store O as bf16 [B][S][H*DK] ----
  const float inv = 1.0f / l_run;
  const size_t obase = ((size_t)(b * S_ + i0 + di)) * D_ + hh * DK_;
#pragma unroll
  for (int dt = 0; dt < 4; ++dt) {
    uint4 w;
    w.x = pk2bf(oc[dt][0] * inv, oc[dt][1] * inv);
    w.y = pk2bf(oc[dt][2] * inv, oc[dt][3] * inv);
    w.z = pk2bf(oc[dt][4] * inv, oc[dt][5] * inv);
    w.w = pk2bf(oc[dt][6] * inv, oc[dt][7] * inv);
    *(uint4*)(O + obase + dt * 16 + 8 * h) = w;
  }
}

// ---------------------------------------------------------------------------
extern "C" void kernel_launch(void* const* d_in, const int* in_sizes, int n_in,
                              void* d_out, int out_size, void* d_ws, size_t ws_size,
                              hipStream_t stream) {
  (void)in_sizes; (void)n_in; (void)out_size; (void)ws_size;
  const float* q  = (const float*)d_in[0];
  const float* k  = (const float*)d_in[1];
  const float* v  = (const float*)d_in[2];
  // d_in[3] = mask (computed analytically in-kernel)
  const float* u  = (const float*)d_in[4];
  const float* vb = (const float*)d_in[5];
  const float* Wq = (const float*)d_in[6];
  const float* Wk = (const float*)d_in[7];
  const float* Wv = (const float*)d_in[8];
  const float* Wr = (const float*)d_in[9];
  const float* Wo = (const float*)d_in[10];
  const float* R  = (const float*)d_in[11];
  float* out = (float*)d_out;

  // Workspace layout (bf16 elements), ~114 MiB total.
  unsigned short* ws  = (unsigned short*)d_ws;
  unsigned short* QU  = ws;                                    // [B][H][S][DK]
  unsigned short* QV  = QU  + (size_t)B_ * H_ * S_ * DK_;      // [B][H][S][DK]
  unsigned short* KH  = QV  + (size_t)B_ * H_ * S_ * DK_;      // [B][H][T][DK]
  unsigned short* VHT = KH  + (size_t)B_ * H_ * T_ * DK_;      // [B][H][DK][T]
  unsigned short* RH  = VHT + (size_t)B_ * H_ * T_ * DK_;      // [H][T][DK]
  unsigned short* O   = RH  + (size_t)H_ * T_ * DK_;           // [B][S][D]
  unsigned short* QB  = O   + (size_t)B_ * S_ * D_;            // bf16 mirrors:
  unsigned short* KB  = QB  + (size_t)B_ * S_ * D_;
  unsigned short* VB  = KB  + (size_t)B_ * T_ * D_;
  unsigned short* RB  = VB  + (size_t)B_ * T_ * D_;
  unsigned short* WQB = RB  + (size_t)T_ * D_;
  unsigned short* WKB = WQB + (size_t)D_ * D_;
  unsigned short* WVB = WKB + (size_t)D_ * D_;
  unsigned short* WRB = WVB + (size_t)D_ * D_;
  unsigned short* WOB = WRB + (size_t)D_ * D_;

  // Stage 0: one-shot f32 -> bf16 conversions (memory-bound, tiny).
  const int nq = B_ * S_ * D_ / 8, nk = B_ * T_ * D_ / 8;
  const int nr = T_ * D_ / 8,      nw = D_ * D_ / 8;
  cvt_bf16_kernel<<<(nq + 255) / 256, 256, 0, stream>>>(q,  QB,  nq);
  cvt_bf16_kernel<<<(nk + 255) / 256, 256, 0, stream>>>(k,  KB,  nk);
  cvt_bf16_kernel<<<(nk + 255) / 256, 256, 0, stream>>>(v,  VB,  nk);
  cvt_bf16_kernel<<<(nr + 255) / 256, 256, 0, stream>>>(R,  RB,  nr);
  cvt_bf16_kernel<<<(nw + 255) / 256, 256, 0, stream>>>(Wq, WQB, nw);
  cvt_bf16_kernel<<<(nw + 255) / 256, 256, 0, stream>>>(Wk, WKB, nw);
  cvt_bf16_kernel<<<(nw + 255) / 256, 256, 0, stream>>>(Wv, WVB, nw);
  cvt_bf16_kernel<<<(nw + 255) / 256, 256, 0, stream>>>(Wr, WRB, nw);
  cvt_bf16_kernel<<<(nw + 255) / 256, 256, 0, stream>>>(Wo, WOB, nw);

  // Stage 1: projections (fused bias+scale / head-split / V-transpose).
  gemm_nt_kernel<EP_QUQV><<<dim3(8, (B_ * S_) / 128), 256, 0, stream>>>(
      QB, WQB, u, vb, QU, QV, nullptr);
  gemm_nt_kernel<EP_K><<<dim3(8, (B_ * T_) / 128), 256, 0, stream>>>(
      KB, WKB, nullptr, nullptr, KH, nullptr, nullptr);
  gemm_nt_kernel<EP_V><<<dim3(8, (B_ * T_) / 128), 256, 0, stream>>>(
      VB, WVB, nullptr, nullptr, VHT, nullptr, nullptr);
  gemm_nt_kernel<EP_R><<<dim3(8, T_ / 128), 256, 0, stream>>>(
      RB, WRB, nullptr, nullptr, RH, nullptr, nullptr);

  // Stage 2: fused flash attention (rel-pos + mask + online softmax).
  attn_kernel<<<dim3(S_ / 128, B_ * H_), 256, 0, stream>>>(QU, QV, KH, VHT, RH, O);

  // Stage 3: output projection: d_out = O @ Wo^T (f32 result).
  gemm_nt_kernel<EP_OUT><<<dim3(8, (B_ * S_) / 128), 256, 0, stream>>>(
      O, WOB, nullptr, nullptr, nullptr, nullptr, out);
}
